// Decoder_59064390255019
// MI455X (gfx1250) — compile-verified, matrix operand order checked by reading, not run on hardware
//
#include <hip/hip_runtime.h>
#include <hip/hip_bf16.h>

// ---------------------------------------------------------------------------
// Decoder LSTM, restructured for MI455X (gfx1250):
//   1) all embedding-side gate contributions precomputed as one WMMA GEMM
//   2) serial h-recurrence done by 4 persistent blocks (16 batch rows each,
//      h in LDS, c in VGPRs, gates in WMMA f32 accumulators)
//   3) output projection c @ W_out^T as one large f16-WMMA GEMM (dominant),
//      software-pipelined so WMMAs overlap the next k-step's loads.
// ---------------------------------------------------------------------------

typedef __attribute__((ext_vector_type(16))) _Float16 v16h;
typedef __attribute__((ext_vector_type(8)))  float    v8f;

union Frag16 {
    v16h v;
    uint4 q[2];
};

// Load a 16x32 f16 A/B fragment: p must already point at
//   base + row*ldk + k0 + ((lane>>4)<<3)
// ISA layout (CDNA5 ISA, 16-bit A 16x32): lane<16 holds K=k0+0..7 and
// k0+16..23; lane>=16 holds K=k0+8..15 and k0+24..31 for row (lane&15).
__device__ __forceinline__ v16h ld_frag(const _Float16* __restrict__ p) {
    Frag16 f;
    f.q[0] = *(const uint4*)(p);
    f.q[1] = *(const uint4*)(p + 16);
    return f.v;
}

__device__ __forceinline__ v16h load_frag(const _Float16* __restrict__ base,
                                          int ldk, int row, int k0, int lane) {
    return ld_frag(base + (size_t)row * ldk + (k0 + ((lane >> 4) << 3)));
}

__device__ __forceinline__ float sigmoid_f(float x) {
    return 1.0f / (1.0f + __expf(-x));
}
__device__ __forceinline__ float tanh_f(float x) {
    x = fminf(15.0f, fmaxf(-15.0f, x));
    float e = __expf(2.0f * x);
    return (e - 1.0f) / (e + 1.0f);
}

#define WMMA_F16(a, b, c) \
    __builtin_amdgcn_wmma_f32_16x16x32_f16(false, (a), false, (b), (short)0, (c), false, false)

// ---------------------------------------------------------------------------
// Generic GEMM: C[M,N] = A[M,K] (f16, row-major) x B[N,K]^T (f16, row-major)
//               + bias[N], fp32 accumulate/output.
// Block = 256 threads = 8 waves. Block tile = 256(M) x 64(N); each wave owns
// a 32(M) x 64(N) strip -> 8 WMMA accumulators, 6 fragment loads per k-step,
// software-pipelined one k-step ahead.
// grid = (N/64, M/256); requires K % 32 == 0, K >= 64.
// ---------------------------------------------------------------------------
__global__ __launch_bounds__(256) void gemm_f16_wmma(
    const _Float16* __restrict__ A, const _Float16* __restrict__ B,
    const float* __restrict__ bias, float* __restrict__ C,
    int M, int N, int K) {
    const int lane = threadIdx.x & 31;
    const int wave = threadIdx.x >> 5;
    const int m0 = blockIdx.y * 256 + wave * 32;
    const int n0 = blockIdx.x * 64;
    const int ln = lane & 15;
    const int koff = (lane >> 4) << 3;

    // running fragment pointers (advance by 32 halves per k-step)
    const _Float16* pa0 = A + (size_t)(m0 + ln) * K + koff;
    const _Float16* pa1 = pa0 + (size_t)16 * K;
    const _Float16* pb0 = B + (size_t)(n0 + ln) * K + koff;
    const _Float16* pb1 = pb0 + (size_t)16 * K;
    const _Float16* pb2 = pb0 + (size_t)32 * K;
    const _Float16* pb3 = pb0 + (size_t)48 * K;

    v8f acc[2][4];
#pragma unroll
    for (int mt = 0; mt < 2; ++mt)
#pragma unroll
        for (int nt = 0; nt < 4; ++nt)
            acc[mt][nt] = v8f{};

    // prologue: fragments for k0 = 0
    v16h a0 = ld_frag(pa0), a1 = ld_frag(pa1);
    v16h b0 = ld_frag(pb0), b1 = ld_frag(pb1), b2 = ld_frag(pb2), b3 = ld_frag(pb3);

    for (int k0 = 0; k0 + 32 < K; k0 += 32) {
        pa0 += 32; pa1 += 32; pb0 += 32; pb1 += 32; pb2 += 32; pb3 += 32;
        // issue next k-step's loads first so WMMAs below overlap them
        v16h na0 = ld_frag(pa0), na1 = ld_frag(pa1);
        v16h nb0 = ld_frag(pb0), nb1 = ld_frag(pb1);
        v16h nb2 = ld_frag(pb2), nb3 = ld_frag(pb3);
        __builtin_prefetch(pb0 + 32, 0, 3);   // k+2 hint -> global_prefetch_b8
        __builtin_prefetch(pb2 + 32, 0, 3);

        acc[0][0] = WMMA_F16(a0, b0, acc[0][0]);
        acc[0][1] = WMMA_F16(a0, b1, acc[0][1]);
        acc[0][2] = WMMA_F16(a0, b2, acc[0][2]);
        acc[0][3] = WMMA_F16(a0, b3, acc[0][3]);
        acc[1][0] = WMMA_F16(a1, b0, acc[1][0]);
        acc[1][1] = WMMA_F16(a1, b1, acc[1][1]);
        acc[1][2] = WMMA_F16(a1, b2, acc[1][2]);
        acc[1][3] = WMMA_F16(a1, b3, acc[1][3]);

        a0 = na0; a1 = na1;
        b0 = nb0; b1 = nb1; b2 = nb2; b3 = nb3;
    }
    // epilogue k-step
    acc[0][0] = WMMA_F16(a0, b0, acc[0][0]);
    acc[0][1] = WMMA_F16(a0, b1, acc[0][1]);
    acc[0][2] = WMMA_F16(a0, b2, acc[0][2]);
    acc[0][3] = WMMA_F16(a0, b3, acc[0][3]);
    acc[1][0] = WMMA_F16(a1, b0, acc[1][0]);
    acc[1][1] = WMMA_F16(a1, b1, acc[1][1]);
    acc[1][2] = WMMA_F16(a1, b2, acc[1][2]);
    acc[1][3] = WMMA_F16(a1, b3, acc[1][3]);

    // D layout (f32 16x16): VGPR r, lanes 0-15 -> M=r, lanes 16-31 -> M=r+8.
    const int mh = (lane >> 4) * 8;
    const float bi0 = bias[n0 +  0 + ln];
    const float bi1 = bias[n0 + 16 + ln];
    const float bi2 = bias[n0 + 32 + ln];
    const float bi3 = bias[n0 + 48 + ln];
#pragma unroll
    for (int mt = 0; mt < 2; ++mt) {
#pragma unroll
        for (int r = 0; r < 8; ++r) {
            float* crow = C + (size_t)(m0 + mt * 16 + mh + r) * N + n0;
            crow[ 0 + ln] = acc[mt][0][r] + bi0;
            crow[16 + ln] = acc[mt][1][r] + bi1;
            crow[32 + ln] = acc[mt][2][r] + bi2;
            crow[48 + ln] = acc[mt][3][r] + bi3;
        }
    }
}

// ---------------------------------------------------------------------------
// LSTM recurrence. grid = 4 blocks; block b owns batch rows [16b, 16b+16).
// Per step: gates(16x2048) = h(16x512,f16 in LDS) @ W_hh^T via WMMA;
// wave w holds gate columns [64w, 64w+64) of all four gate quarters in 16
// accumulators, so i/f/g/o for a column live in the same wave -> cell update
// fully in registers. c state lives in VGPRs across all 64 steps.
// B fragments are pipelined one gate-group (4 frags) ahead of the WMMAs.
// ---------------------------------------------------------------------------
__global__ __launch_bounds__(256) void lstm_recurrence(
    const float* __restrict__ h_in, const float* __restrict__ c_in,
    const float* __restrict__ pregates,      // [T*64, 2048] fp32 (incl. b_ih+b_hh)
    const _Float16* __restrict__ Whh,        // [2048, 512] f16
    _Float16* __restrict__ c_all,            // [T*64, 512] f16 out
    int T) {
    __shared__ __align__(16) _Float16 hlds[16 * 512];

    const int lane  = threadIdx.x & 31;
    const int wave  = threadIdx.x >> 5;
    const int m0    = blockIdx.x * 16;       // global batch-row base
    const int ln    = lane & 15;
    const int mh    = (lane >> 4) * 8;
    const int jbase = wave * 64;             // per-gate column slice

    // init h (f16) into LDS
    for (int idx = threadIdx.x; idx < 16 * 512; idx += 256)
        hlds[idx] = (_Float16)h_in[(size_t)(m0 + (idx >> 9)) * 512 + (idx & 511)];

    // init c state: cst[u][r] = c[m0+mh+r, jbase + 16u + ln]
    float cst[4][8];
#pragma unroll
    for (int u = 0; u < 4; ++u)
#pragma unroll
        for (int r = 0; r < 8; ++r)
            cst[u][r] = c_in[(size_t)(m0 + mh + r) * 512 + jbase + u * 16 + ln];

    __syncthreads();

    for (int t = 0; t < T; ++t) {
        v8f acc[4][4];
#pragma unroll
        for (int q = 0; q < 4; ++q)
#pragma unroll
            for (int u = 0; u < 4; ++u)
                acc[q][u] = v8f{};

        for (int k0 = 0; k0 < 512; k0 += 32) {
            // A fragment from LDS (ds_load_b128 x2)
            const _Float16* ap = &hlds[((lane & 15) << 9) + k0 + ((lane >> 4) << 3)];
            Frag16 af;
            af.q[0] = *(const uint4*)(ap);
            af.q[1] = *(const uint4*)(ap + 16);

            // pipelined over gate groups: load group q+1 while computing q
            v16h bq[4];
#pragma unroll
            for (int u = 0; u < 4; ++u)
                bq[u] = load_frag(Whh, 512, jbase + u * 16 + ln, k0, lane);
#pragma unroll
            for (int q = 0; q < 4; ++q) {
                v16h bn[4];
                if (q < 3) {
#pragma unroll
                    for (int u = 0; u < 4; ++u)
                        bn[u] = load_frag(Whh, 512,
                                          (q + 1) * 512 + jbase + u * 16 + ln,
                                          k0, lane);
                }
#pragma unroll
                for (int u = 0; u < 4; ++u)
                    acc[q][u] = WMMA_F16(af.v, bq[u], acc[q][u]);
                if (q < 3) {
#pragma unroll
                    for (int u = 0; u < 4; ++u)
                        bq[u] = bn[u];
                }
            }
        }
        __syncthreads();  // everyone done reading old h from LDS

        const float* pg = pregates + (size_t)(t * 64 + m0) * 2048;
#pragma unroll
        for (int u = 0; u < 4; ++u) {
#pragma unroll
            for (int r = 0; r < 8; ++r) {
                const int m = mh + r;
                const int j = jbase + u * 16 + ln;
                const float* pgm = pg + (size_t)m * 2048 + j;
                const float gi = acc[0][u][r] + pgm[0];
                const float gf = acc[1][u][r] + pgm[512];
                const float gg = acc[2][u][r] + pgm[1024];
                const float go = acc[3][u][r] + pgm[1536];
                const float cn = sigmoid_f(gf) * cst[u][r] + sigmoid_f(gi) * tanh_f(gg);
                cst[u][r] = cn;
                const float hn = sigmoid_f(go) * tanh_f(cn);
                hlds[(m << 9) + j] = (_Float16)hn;
                c_all[(size_t)(t * 64 + m0 + m) * 512 + j] = (_Float16)cn;
            }
        }
        __syncthreads();  // new h visible before next step's GEMM
    }
}

// ---------------------------------------------------------------------------
// Prep kernels
// ---------------------------------------------------------------------------
__global__ void cvt_f16(const float* __restrict__ src, _Float16* __restrict__ dst, int n) {
    int i = blockIdx.x * 256 + threadIdx.x;
    if (i < n) dst[i] = (_Float16)src[i];
}

__global__ void bias_comb_k(const float* __restrict__ a, const float* __restrict__ b,
                            float* __restrict__ out, int n) {
    int i = blockIdx.x * 256 + threadIdx.x;
    if (i < n) out[i] = a[i] + b[i];
}

// Teacher-forced token embedding gather -> f16 X[T*B, E]. grid = T*B, block = E.
__global__ void gather_embed(const int* __restrict__ y, const float* __restrict__ embed,
                             _Float16* __restrict__ X) {
    const int m = blockIdx.x;           // m = t*64 + b
    const int e = threadIdx.x;
    const int t = m >> 6, b = m & 63;
    const int tok = (t == 0) ? 1 /*SOS*/ : y[(t - 1) * 64 + b];
    X[(size_t)m * 256 + e] = (_Float16)embed[(size_t)tok * 256 + e];
}

// ---------------------------------------------------------------------------
extern "C" void kernel_launch(void* const* d_in, const int* in_sizes, int n_in,
                              void* d_out, int out_size, void* d_ws, size_t ws_size,
                              hipStream_t stream) {
    (void)in_sizes; (void)n_in; (void)out_size; (void)ws_size;

    const float* h      = (const float*)d_in[0];   // [64,512]
    const float* c      = (const float*)d_in[1];   // [64,512]
    /* d_in[2] = context: unused by reference step */
    const int*   y      = (const int*)d_in[3];     // [64,64]
    const float* embed  = (const float*)d_in[4];   // [32000,256]
    const float* W_ih   = (const float*)d_in[5];   // [2048,256]
    const float* b_ih   = (const float*)d_in[6];   // [2048]
    const float* W_hh   = (const float*)d_in[7];   // [2048,512]
    const float* b_hh   = (const float*)d_in[8];   // [2048]
    const float* W_out  = (const float*)d_in[9];   // [32000,512]
    const float* b_out  = (const float*)d_in[10];  // [32000]
    float* out = (float*)d_out;                    // [64,64,32000]

    constexpr int T = 64, B = 64, E = 256, H = 512, V = 32000;
    constexpr int M = T * B;       // 4096
    constexpr int G = 4 * H;       // 2048

    char* ws = (char*)d_ws;
    size_t off = 0;
    auto alloc = [&](size_t bytes) -> void* {
        void* p = ws + off;
        off = (off + bytes + 255) & ~(size_t)255;
        return p;
    };
    _Float16* Wih_h   = (_Float16*)alloc((size_t)G * E * 2);   // 1.0 MB
    _Float16* Whh_h   = (_Float16*)alloc((size_t)G * H * 2);   // 2.1 MB
    _Float16* Wout_h  = (_Float16*)alloc((size_t)V * H * 2);   // 32.8 MB
    _Float16* Xh      = (_Float16*)alloc((size_t)M * E * 2);   // 2.1 MB
    float*    biasC   = (float*)   alloc((size_t)G * 4);       // 8 KB
    float*    pregate = (float*)   alloc((size_t)M * G * 4);   // 33.6 MB
    _Float16* c_allh  = (_Float16*)alloc((size_t)M * H * 2);   // 4.2 MB

    // 1) precision conversion of weights
    cvt_f16<<<(G * E + 255) / 256, 256, 0, stream>>>(W_ih, Wih_h, G * E);
    cvt_f16<<<(G * H + 255) / 256, 256, 0, stream>>>(W_hh, Whh_h, G * H);
    cvt_f16<<<(V * H + 255) / 256, 256, 0, stream>>>(W_out, Wout_h, V * H);
    bias_comb_k<<<(G + 255) / 256, 256, 0, stream>>>(b_ih, b_hh, biasC, G);

    // 2) teacher-forced embedding gather
    gather_embed<<<M, E, 0, stream>>>(y, embed, Xh);

    // 3) pre-gates for all steps: X @ W_ih^T + (b_ih + b_hh)   [4096 x 2048]
    gemm_f16_wmma<<<dim3(G / 64, M / 256), 256, 0, stream>>>(
        Xh, Wih_h, biasC, pregate, M, G, E);

    // 4) serial recurrence (4 independent batch-slices, one launch)
    lstm_recurrence<<<4, 256, 0, stream>>>(h, c, pregate, Whh_h, c_allh, T);

    // 5) output projection: c_all @ W_out^T + b_out   [4096 x 32000] (dominant)
    gemm_f16_wmma<<<dim3(V / 64, M / 256), 256, 0, stream>>>(
        c_allh, Wout_h, b_out, out, M, V, H);
}
